// BatchedGAT_60146722013278
// MI455X (gfx1250) — matrix operand order repaired and checked by reading, not compiled, not run on hardware
//
#include <hip/hip_runtime.h>
#include <hip/hip_bf16.h>

// ---------------------------------------------------------------------------
// Batched dense-adjacency GAT for MI455X (gfx1250, wave32, WMMA + TDM).
//   B=16, N=1024, DIN=256, HEADS=4, F=64, OUT=256
// Pipeline:
//   k0: W[256][256] -> f16 WT[c][k]            (transpose for B-operand loads)
//   k1: h = x @ W   via v_wmma_f32_16x16x32_f16, stored f16 hT[b*H+h][f][n]
//   k2: e_src/e_dst logits from hT
//   k3: per-row softmax max/sum (online, one coalesced pass over adj)
//   k4: fused  out = softmax(mask(leaky(e))) @ h  via WMMA; adj slabs DMA'd
//       into LDS by the Tensor Data Mover (tensor_load_to_lds + TENSORcnt),
//       with D# pad_enable providing a bank-conflict-free padded layout.
// ---------------------------------------------------------------------------

typedef __attribute__((ext_vector_type(16))) _Float16 v16h;
typedef __attribute__((ext_vector_type(8)))  _Float16 v8h;
typedef __attribute__((ext_vector_type(8)))  float    v8f;
typedef __attribute__((ext_vector_type(4)))  unsigned int u32x4;
typedef __attribute__((ext_vector_type(8)))  int          i32x8;
typedef __attribute__((ext_vector_type(4)))  int          i32x4;

#define B_    16
#define N_    1024
#define DIN_  256
#define H_    4
#define F_    64
#define C_    256        // HEADS*F
#define NEG_SLOPE 0.2f
#define MASK_VAL  (-1e9f)
#define JCHUNK 512       // adj columns staged per LDS slab

#if defined(__HIP_DEVICE_COMPILE__) && __has_builtin(__builtin_amdgcn_tensor_load_to_lds)
#define HAVE_TDM 1
#else
#define HAVE_TDM 0
#endif

// TDM pad layout: 1 DWORD pad inserted after every 256 DWORDs streamed.
// Element (row, col) of a 16 x JCHUNK tile lands at dword  g + (g>>8),
// g = row*JCHUNK + col  ->  row stride 514 dwords -> banks 2*row (distinct).
__device__ __forceinline__ int adj_idx(int row, int col) {
    int g = (row << 9) + col;     // JCHUNK == 512
    return g + (g >> 8);
}

// ---------------------------------------------------------------- kernel 0
// WT[c][k] = (f16) W[k][c]   (W flat layout [DIN][H][F] == [k][c])
__global__ __launch_bounds__(256)
void gat_wt_kernel(const float* __restrict__ W, _Float16* __restrict__ WT) {
    int idx = blockIdx.x * 256 + threadIdx.x;    // 65536 total
    int k = idx & 255;
    int c = idx >> 8;
    WT[c * DIN_ + k] = (_Float16)W[k * C_ + c];
}

// ---------------------------------------------------------------- kernel 1
// One wave computes a 16(n) x 16(c) tile of h = x @ W, K-loop over 256.
// A-layout (f16 16x32): lane<16 row=lane, K = {0..15}+16*lg (contiguous, same order as B).
// Output stored transposed: hT[(b*C + c)*N + n] (f16), 16B-aligned v8h store.
__global__ __launch_bounds__(32)
void gat_gemm_h_kernel(const float* __restrict__ x,
                       const _Float16* __restrict__ WT,
                       _Float16* __restrict__ hT) {
    int bid = blockIdx.x;                 // B * (N/16) * (C/16) = 16384
    int c0 = (bid & 15) << 4;
    int n0 = ((bid >> 4) & 63) << 4;
    int b  = bid >> 10;
    int lane = threadIdx.x;
    int lg = lane >> 4;                   // lane group 0/1
    int ll = lane & 15;

    const float*    xrow = x  + ((size_t)b * N_ + (n0 + ll)) * DIN_;
    const _Float16* wrow = WT + (size_t)(c0 + ll) * DIN_;

    v8f acc = {};
    for (int k0 = 0; k0 < DIN_; k0 += 32) {
        int ka = k0 + 16 * lg;            // this lane's first K octet
        float4 xa = *(const float4*)(xrow + ka);
        float4 xb = *(const float4*)(xrow + ka + 4);
        float4 xc = *(const float4*)(xrow + ka + 8);
        float4 xd = *(const float4*)(xrow + ka + 12);
        v16h a;
        a[0]=(_Float16)xa.x; a[1]=(_Float16)xa.y; a[2]=(_Float16)xa.z; a[3]=(_Float16)xa.w;
        a[4]=(_Float16)xb.x; a[5]=(_Float16)xb.y; a[6]=(_Float16)xb.z; a[7]=(_Float16)xb.w;
        a[8]=(_Float16)xc.x; a[9]=(_Float16)xc.y; a[10]=(_Float16)xc.z; a[11]=(_Float16)xc.w;
        a[12]=(_Float16)xd.x; a[13]=(_Float16)xd.y; a[14]=(_Float16)xd.z; a[15]=(_Float16)xd.w;
        // B-layout (f16 32x16): lane col = ll, K = {0..15} + 16*lg, contiguous in WT row
        v16h bm = *(const v16h*)(wrow + k0 + 16 * lg);
        acc = __builtin_amdgcn_wmma_f32_16x16x32_f16(false, a, false, bm,
                                                     (short)0, acc, false, false);
    }
    // C-layout: acc[v] is row M = v + 8*lg, col = ll. 8 consecutive n -> v8h store.
    v8h hout;
    #pragma unroll
    for (int v = 0; v < 8; ++v) hout[v] = (_Float16)acc[v];
    _Float16* dst = hT + ((size_t)(b * C_ + c0 + ll)) * N_ + n0 + 8 * lg;
    *(v8h*)dst = hout;
}

// ---------------------------------------------------------------- kernel 2
// e_src[bh][n] = sum_f hT[bh][f][n] * a_src[h][f];  same for e_dst.
__global__ __launch_bounds__(256)
void gat_logits_kernel(const _Float16* __restrict__ hT,
                       const float* __restrict__ a_src,
                       const float* __restrict__ a_dst,
                       float* __restrict__ e_src,
                       float* __restrict__ e_dst) {
    int idx = blockIdx.x * 256 + threadIdx.x;    // B*H*N = 65536
    int n  = idx & (N_ - 1);
    int bh = idx >> 10;
    int h  = bh & (H_ - 1);
    const _Float16* base = hT + (size_t)bh * F_ * N_ + n;
    float s = 0.f, d = 0.f;
    #pragma unroll 8
    for (int f = 0; f < F_; ++f) {
        float hv = (float)base[(size_t)f * N_];
        s += hv * a_src[h * F_ + f];
        d += hv * a_dst[h * F_ + f];
    }
    e_src[idx] = s;
    e_dst[idx] = d;
}

// ---------------------------------------------------------------- kernel 3
// One wave per (b,i): online softmax max/sum over j for all 4 heads,
// single coalesced pass over adj row, shuffle-merged across 32 lanes.
__global__ __launch_bounds__(32)
void gat_rowstats_kernel(const float* __restrict__ adj,
                         const float* __restrict__ e_src,
                         const float* __restrict__ e_dst,
                         float* __restrict__ rowmax,
                         float* __restrict__ rowsum) {
    int bid = blockIdx.x;                 // B*N = 16384
    int i = bid & (N_ - 1);
    int b = bid >> 10;
    int lane = threadIdx.x;

    const float* arow = adj + ((size_t)b * N_ + i) * N_;
    float m[H_], s[H_], ed[H_];
    #pragma unroll
    for (int h = 0; h < H_; ++h) {
        m[h] = -INFINITY; s[h] = 0.f;
        ed[h] = e_dst[(b * H_ + h) * N_ + i];
    }
    for (int t = 0; t < N_ / 32; ++t) {
        int j = t * 32 + lane;
        float av = arow[j];
        bool edge = av > 0.5f;
        #pragma unroll
        for (int h = 0; h < H_; ++h) {
            float e = ed[h] + e_src[(b * H_ + h) * N_ + j];
            e = fmaxf(e, 0.f) + NEG_SLOPE * fminf(e, 0.f);   // leaky relu
            e = edge ? e : MASK_VAL;
            float mn = fmaxf(m[h], e);
            s[h] = s[h] * __expf(m[h] - mn) + __expf(e - mn);
            m[h] = mn;
        }
    }
    #pragma unroll
    for (int off = 16; off >= 1; off >>= 1) {
        #pragma unroll
        for (int h = 0; h < H_; ++h) {
            float mo = __shfl_xor(m[h], off, 32);
            float so = __shfl_xor(s[h], off, 32);
            float mn = fmaxf(m[h], mo);
            s[h] = s[h] * __expf(m[h] - mn) + so * __expf(mo - mn);
            m[h] = mn;
        }
    }
    if (lane == 0) {
        #pragma unroll
        for (int h = 0; h < H_; ++h) {
            rowmax[(b * H_ + h) * N_ + i] = m[h];
            rowsum[(b * H_ + h) * N_ + i] = s[h];
        }
    }
}

// ---------------------------------------------------------------- kernel 4
// WG = (b, 16-row i-tile); 4 waves = 4 heads.
//  * e_src for all heads staged once in LDS (16 KB).
//  * adj staged in 16 x 512 slabs: one tensor_load_to_lds DMA per slab
//    (TENSORcnt) with D# padding -> conflict-free 514-dword row stride.
//  * j-loop: build f16 A-tile of exp(masked leaky logits - rowmax) in WMMA
//    A layout, 4 x v_wmma_f32_16x16x32_f16 per 32-wide j-block, no barriers.
//  * Epilogue: rows * 1/rowsum (exact softmax; all-masked row -> 1/N) + bias.
__global__ __launch_bounds__(128)
void gat_agg_kernel(const float* __restrict__ adj,
                    const _Float16* __restrict__ hT,
                    const float* __restrict__ e_src,
                    const float* __restrict__ e_dst,
                    const float* __restrict__ rowmax,
                    const float* __restrict__ rowsum,
                    const float* __restrict__ bias,
                    float* __restrict__ out) {
    __shared__ float adjLds[16 * (JCHUNK + 2)];   // 16 x 514 dwords (TDM pad layout)
    __shared__ float esLds[H_ * N_];              // e_src, all heads (16 KB)

    int bid = blockIdx.x;                 // B * (N/16) = 1024
    int i0 = (bid & 63) << 4;
    int b  = bid >> 6;
    int tid  = threadIdx.x;
    int hd   = tid >> 5;                  // wave id == head
    int lane = tid & 31;
    int lg = lane >> 4, ll = lane & 15;

    int bh = b * H_ + hd;
    float ed = e_dst[bh * N_ + i0 + ll];  // this lane's row logit
    float mi = rowmax[bh * N_ + i0 + ll]; // this lane's row max

    const float*    adjbase = adj + (size_t)b * N_ * N_;
    const _Float16* hbase   = hT + (size_t)bh * F_ * N_;

    // Stage e_src for all 4 heads once (coalesced).
    #pragma unroll
    for (int it = 0; it < 32; ++it) {
        int idx = it * 128 + tid;         // 4096 floats
        esLds[idx] = e_src[(b * H_ + (idx >> 10)) * N_ + (idx & (N_ - 1))];
    }

    v8f c0 = {}, c1 = {}, c2 = {}, c3 = {};

    for (int jc = 0; jc < N_; jc += JCHUNK) {
        __syncthreads();                  // previous slab's readers done
#if HAVE_TDM
        if (hd == 0) {
            // Tensor DMA: 16 x 512 f32 tile of adj -> padded LDS slab.
            uint64_t ga = (uint64_t)(uintptr_t)(adjbase + (size_t)i0 * N_ + jc);
            uint32_t ldsOff = (uint32_t)(uintptr_t)(void*)&adjLds[0];
            u32x4 g0 = {};
            g0.x = 1u;                                    // count=1, user mode
            g0.y = ldsOff;                                // lds_addr (bytes)
            g0.z = (uint32_t)ga;                          // global_addr[31:0]
            g0.w = (uint32_t)((ga >> 32) & 0x01FFFFFFu)   // global_addr[56:32]
                 | (2u << 30);                            // type = 2 ("image")
            i32x8 g1 = {};
            g1[0] = (2 << 16)        // data_size = 4 bytes
                  | (1 << 20)        // pad_enable
                  | (7 << 22);       // pad_interval: every 256 DWORDs, amount=1 DWORD
            g1[1] = (int)((N_ & 0xFFFF) << 16);                  // tensor_dim0 lo16
            g1[2] = (int)(((N_ >> 16) & 0xFFFF) | ((N_ & 0xFFFF) << 16)); // dim0 hi | dim1 lo
            g1[3] = (int)(((N_ >> 16) & 0xFFFF) | (JCHUNK << 16));        // dim1 hi | tile_dim0
            g1[4] = 16;                                          // tile_dim1 (tile_dim2=0)
            g1[5] = N_;                                          // tensor_dim0_stride lo32
            g1[6] = 0;                                           // stride hi16 | dim1_stride lo16
            g1[7] = 0;
            i32x4 gz4 = {};                                      // groups 2/3: unused (2D tile)
            i32x8 gz8 = {};                                      // trailing desc words: zero
            __builtin_amdgcn_tensor_load_to_lds(g0, g1, gz4, gz4, gz8, 0);
            __builtin_amdgcn_s_wait_tensorcnt(0);
        }
#else
        // Fallback: cooperative synchronous staging into the same padded layout.
        #pragma unroll
        for (int it = 0; it < 16; ++it) {                 // 2048 float4 / 128 thr
            int id  = it * 128 + tid;
            int row = id >> 7;                            // 128 float4 per row
            int c4  = (id & 127) << 2;
            float4 av = *(const float4*)(adjbase + (size_t)(i0 + row) * N_ + jc + c4);
            int base = adj_idx(row, c4);                  // c4..c4+3 stay in one pad block
            adjLds[base + 0] = av.x; adjLds[base + 1] = av.y;
            adjLds[base + 2] = av.z; adjLds[base + 3] = av.w;
        }
#endif
        __syncthreads();                  // slab visible to all waves

        for (int j0 = 0; j0 < JCHUNK; j0 += 32) {
            // A-tile in WMMA f16 16x32 layout:
            //   lane<16: row=ll, K = hv; lane>=16: +16 more.
            v16h a;
            #pragma unroll
            for (int hv = 0; hv < 16; ++hv) {
                int K = hv + (lg << 4);
                int jcol = j0 + K;
                float e = ed + esLds[hd * N_ + jc + jcol];
                e = fmaxf(e, 0.f) + NEG_SLOPE * fminf(e, 0.f);
                e = (adjLds[adj_idx(ll, jcol)] > 0.5f) ? e : MASK_VAL;
                a[hv] = (_Float16)__expf(e - mi);     // in [0,1]
            }

            // B tiles: lane col = ll, K = {0..15}+16*lg contiguous (32B-aligned v16h)
            const _Float16* hb = hbase + jc + j0 + (lg << 4);
            v16h b0 = *(const v16h*)(hb + (size_t)(ll     ) * N_);
            v16h b1 = *(const v16h*)(hb + (size_t)(ll + 16) * N_);
            v16h b2 = *(const v16h*)(hb + (size_t)(ll + 32) * N_);
            v16h b3 = *(const v16h*)(hb + (size_t)(ll + 48) * N_);

            c0 = __builtin_amdgcn_wmma_f32_16x16x32_f16(false, a, false, b0, (short)0, c0, false, false);
            c1 = __builtin_amdgcn_wmma_f32_16x16x32_f16(false, a, false, b1, (short)0, c1, false, false);
            c2 = __builtin_amdgcn_wmma_f32_16x16x32_f16(false, a, false, b2, (short)0, c2, false, false);
            c3 = __builtin_amdgcn_wmma_f32_16x16x32_f16(false, a, false, b3, (short)0, c3, false, false);
        }
    }

    // Epilogue: row-scale by 1/rowsum (exact softmax), + bias.
    #pragma unroll
    for (int v = 0; v < 8; ++v) {
        int M = v + (lg << 3);
        float inv_s = 1.0f / rowsum[bh * N_ + i0 + M];
        float* orow = out + ((size_t)b * N_ + i0 + M) * C_ + hd * F_;
        orow[ll     ] = c0[v] * inv_s + bias[hd * F_ + ll     ];
        orow[ll + 16] = c1[v] * inv_s + bias[hd * F_ + ll + 16];
        orow[ll + 32] = c2[v] * inv_s + bias[hd * F_ + ll + 32];
        orow[ll + 48] = c3[v] * inv_s + bias[hd * F_ + ll + 48];
    }
}

// ---------------------------------------------------------------------------
extern "C" void kernel_launch(void* const* d_in, const int* in_sizes, int n_in,
                              void* d_out, int out_size, void* d_ws, size_t ws_size,
                              hipStream_t stream) {
    const float* x     = (const float*)d_in[0];   // [B,N,DIN]
    const float* adj   = (const float*)d_in[1];   // [B,N,N]
    const float* W     = (const float*)d_in[2];   // [DIN,H,F]
    const float* a_src = (const float*)d_in[3];   // [H,F]
    const float* a_dst = (const float*)d_in[4];   // [H,F]
    const float* bias  = (const float*)d_in[5];   // [C]
    float* out = (float*)d_out;                   // [B,N,C]

    // Workspace carve-up (all offsets well aligned).
    char* ws = (char*)d_ws;
    size_t off = 0;
    _Float16* WT = (_Float16*)(ws + off); off += (size_t)C_ * DIN_ * 2;          // 128 KB
    _Float16* hT = (_Float16*)(ws + off); off += (size_t)B_ * C_ * N_ * 2;       // 8 MB
    float* e_src = (float*)(ws + off);    off += (size_t)B_ * H_ * N_ * 4;       // 256 KB
    float* e_dst = (float*)(ws + off);    off += (size_t)B_ * H_ * N_ * 4;
    float* rmax  = (float*)(ws + off);    off += (size_t)B_ * H_ * N_ * 4;
    float* rsum  = (float*)(ws + off);    off += (size_t)B_ * H_ * N_ * 4;
    (void)ws_size; (void)in_sizes; (void)n_in; (void)out_size;

    gat_wt_kernel      <<<256,   256, 0, stream>>>(W, WT);
    gat_gemm_h_kernel  <<<16384,  32, 0, stream>>>(x, WT, hT);
    gat_logits_kernel  <<<256,   256, 0, stream>>>(hT, a_src, a_dst, e_src, e_dst);
    gat_rowstats_kernel<<<16384,  32, 0, stream>>>(adj, e_src, e_dst, rmax, rsum);
    gat_agg_kernel     <<<1024,  128, 0, stream>>>(adj, hT, e_src, e_dst, rmax, rsum, bias, out);
}
